// DeepEMD_575525617691
// MI455X (gfx1250) — matrix operand order repaired and checked by reading, not compiled, not run on hardware
//
#include <hip/hip_runtime.h>
#include <hip/hip_bf16.h>

// ---------------------------------------------------------------------------
// DeepEMD fused pipeline for gfx1250 (MI455X), wave32 + WMMA bf16.
//   query: [8, 75, 640, 8, 8] f32     proto: [8, 5, 640, 8, 8] f32
//   out  : [8, 75, 5] f32 logits
// ---------------------------------------------------------------------------

#define B_  8
#define Q_  75
#define W_  5
#define C_  640
#define HW_ 64
#define NP_ (B_*Q_*W_)     // 3000 independent EMD problems
#define SINK_EPS   0.05f
#define INV_EPS    20.0f
#define SINK_ITERS 50
#define TEMP_      12.5f

typedef __attribute__((ext_vector_type(16))) __bf16 v16bf;
typedef __attribute__((ext_vector_type(8)))  float  v8f;

union FragU { uint4 u[2]; v16bf v; };

static __device__ __forceinline__ unsigned short f2bf(float x) {
    unsigned int u = __float_as_uint(x);
    unsigned int r = (u + 0x7FFFu + ((u >> 16) & 1u)) >> 16;   // RNE
    return (unsigned short)r;
}
static __device__ __forceinline__ float bf2f(unsigned short h) {
    return __uint_as_float(((unsigned int)h) << 16);
}

// ---------------------------------------------------------------------------
// Kernel 1: per (b,m): channel-L2 normalize, write bf16 [site][c] (transposed)
// and f32 pooled vector (mean over sites of normalized features).
// src layout: [bm][c][site]  (c-major over 64 sites)
// ---------------------------------------------------------------------------
__global__ __launch_bounds__(256)
void norm_pool_kernel(const float* __restrict__ src,
                      unsigned short* __restrict__ dstn,
                      float* __restrict__ pool) {
    const int bm = blockIdx.x;
    const int t  = threadIdx.x;
    __shared__ float red[256];
    __shared__ float inv[HW_];
    __shared__ float tile[64 * 65];

    const float* sp = src + (size_t)bm * C_ * HW_;

    // squared-norm per site, 4 threads per site over strided c
    {
        const int x = t & 63, c0 = t >> 6;
        float ss = 0.f;
        for (int cc = c0; cc < C_; cc += 4) {
            float v = sp[cc * HW_ + x];
            ss += v * v;
        }
        red[t] = ss;
    }
    __syncthreads();
    if (t < HW_) {
        float s = red[t] + red[t + 64] + red[t + 128] + red[t + 192];
        inv[t] = 1.0f / fmaxf(sqrtf(s), 1e-8f);
    }
    __syncthreads();

    // pooled normalized vector (f32)
    float* pp = pool + (size_t)bm * C_;
    for (int cc = t; cc < C_; cc += 256) {
        const float* row = sp + (size_t)cc * HW_;
        float s = 0.f;
        #pragma unroll 8
        for (int xx = 0; xx < HW_; ++xx) s += row[xx] * inv[xx];
        pp[cc] = s * (1.0f / 64.0f);
    }

    // normalized bf16, transposed to [site][c] via LDS tiles (coalesced both ways)
    unsigned short* dp = dstn + (size_t)bm * HW_ * C_;
    for (int tc = 0; tc < C_; tc += 64) {
        __syncthreads();
        for (int i = t; i < 64 * 64; i += 256) {
            int xx = i & 63, ccl = i >> 6;
            tile[ccl * 65 + xx] = sp[(size_t)(tc + ccl) * HW_ + xx] * inv[xx];
        }
        __syncthreads();
        for (int i = t; i < 64 * 64; i += 256) {
            int ccl = i & 63, xx = i >> 6;
            dp[(size_t)xx * C_ + tc + ccl] = f2bf(tile[ccl * 65 + xx]);
        }
    }
}

// ---------------------------------------------------------------------------
// Kernel 2: per (b,q,w): weight vectors -> log row/col masses.
//   w1[x] = relu(<qn[b,q,x,:], ppool[b,w,:]>) + 1e-3  -> loga = log(w1/sum)
//   w2[y] = relu(<pn[b,w,y,:], qpool[b,q,:]>) + 1e-3  -> logb = log(w2/sum)
// ---------------------------------------------------------------------------
__device__ __forceinline__ float blockSum64(float* r, float v, int x) {
    r[x] = v; __syncthreads();
    for (int s = 32; s > 0; s >>= 1) { if (x < s) r[x] += r[x + s]; __syncthreads(); }
    float res = r[0];
    __syncthreads();
    return res;
}

__global__ __launch_bounds__(64)
void weights_kernel(const unsigned short* __restrict__ qn,
                    const unsigned short* __restrict__ pn,
                    const float* __restrict__ qpool,
                    const float* __restrict__ ppool,
                    float* __restrict__ loga,
                    float* __restrict__ logb) {
    const int pid = blockIdx.x;
    const int b = pid / (Q_ * W_);
    const int rm = pid % (Q_ * W_);
    const int q = rm / W_;
    const int w = rm % W_;
    const int x = threadIdx.x;
    __shared__ float r[64];

    const unsigned short* qrow = qn + ((size_t)(b * Q_ + q) * HW_ + x) * C_;
    const float*          ppv  = ppool + (size_t)(b * W_ + w) * C_;
    float d1 = 0.f;
    #pragma unroll 8
    for (int c = 0; c < C_; ++c) d1 += bf2f(qrow[c]) * ppv[c];
    float w1 = fmaxf(d1, 0.f) + 0.001f;
    float s1 = blockSum64(r, w1, x);
    loga[(size_t)pid * HW_ + x] = __logf(w1) - __logf(s1);

    const unsigned short* prow = pn + ((size_t)(b * W_ + w) * HW_ + x) * C_;
    const float*          qpv  = qpool + (size_t)(b * Q_ + q) * C_;
    float d2 = 0.f;
    #pragma unroll 8
    for (int c = 0; c < C_; ++c) d2 += bf2f(prow[c]) * qpv[c];
    float w2 = fmaxf(d2, 0.f) + 0.001f;
    float s2 = blockSum64(r, w2, x);
    logb[(size_t)pid * HW_ + x] = __logf(w2) - __logf(s2);
}

// ---------------------------------------------------------------------------
// Kernel 3: one workgroup (128 thr = 4 waves) per (b,q,w) EMD problem.
//   Phase A: sim = qn_x . pn_y  via v_wmma_f32_16x16x32_bf16 (K=640 loop) -> LDS
//   Phase B: 50 Sinkhorn iterations in LDS (log-domain, max-stabilized LSE)
//   Phase C: logits = sum(sim * plan) * (T/n)
// ---------------------------------------------------------------------------
__global__ __launch_bounds__(128)
void emd_kernel(const unsigned short* __restrict__ qn,
                const unsigned short* __restrict__ pn,
                const float* __restrict__ loga,
                const float* __restrict__ logb,
                float* __restrict__ out) {
    const int pid = blockIdx.x;
    const int b = pid / (Q_ * W_);
    const int rm = pid % (Q_ * W_);
    const int q = rm / W_;
    const int w = rm % W_;

    __shared__ float sim[HW_ * HW_];          // 16 KB
    __shared__ float fv[HW_], gv[HW_], la[HW_], lb[HW_];
    __shared__ float red[128];

    const int t  = threadIdx.x;
    const int wv = t >> 5;        // wave 0..3 -> 16-row band
    const int l  = t & 31;        // lane
    const int lr = l & 15;
    const int hi = l >> 4;
    const int mBase = wv * 16;

    const unsigned short* qbase = qn + (size_t)(b * Q_ + q) * HW_ * C_;
    const unsigned short* pbase = pn + (size_t)(b * W_ + w) * HW_ * C_;

    // --- Phase A: WMMA GEMM, each wave computes a 16x64 band of sim ---
    const v8f zero = {0.f, 0.f, 0.f, 0.f, 0.f, 0.f, 0.f, 0.f};
    v8f acc[4] = {zero, zero, zero, zero};

    const unsigned short* qrow = qbase + (size_t)(mBase + lr) * C_;
    const int koffA = hi ? 8 : 0;     // A frag: lanes 0-15 K 0-7/16-23, lanes 16-31 K 8-15/24-31
    const int koffB = hi ? 8 : 0;     // B frag: lanes 0-15 K 0-7/16-23, lanes 16-31 K 8-15/24-31

    for (int kb = 0; kb < C_; kb += 32) {
        FragU af;
        af.u[0] = *(const uint4*)(qrow + kb + koffA);
        af.u[1] = *(const uint4*)(qrow + kb + koffA + 16);
        __builtin_prefetch(qrow + kb + koffA + 32, 0, 0);
        #pragma unroll
        for (int nt = 0; nt < 4; ++nt) {
            const unsigned short* prow = pbase + (size_t)(nt * 16 + lr) * C_;
            FragU bf;
            bf.u[0] = *(const uint4*)(prow + kb + koffB);
            bf.u[1] = *(const uint4*)(prow + kb + koffB + 16);
            acc[nt] = __builtin_amdgcn_wmma_f32_16x16x32_bf16(
                false, af.v, false, bf.v, (short)0, acc[nt], false, false);
        }
    }

    // C/D layout: lane holds N = l&15; VGPR r holds M = r + 8*(l>>4)
    #pragma unroll
    for (int nt = 0; nt < 4; ++nt) {
        #pragma unroll
        for (int r = 0; r < 8; ++r) {
            sim[(mBase + r + 8 * hi) * HW_ + nt * 16 + lr] = acc[nt][r];
        }
    }

    if (t < HW_) { la[t] = loga[(size_t)pid * HW_ + t]; fv[t] = 0.f; }
    else         { int j = t - HW_; lb[j] = logb[(size_t)pid * HW_ + j]; gv[j] = 0.f; }
    __syncthreads();

    // --- Phase B: Sinkhorn (cost = 1 - sim) ---
    for (int it = 0; it < SINK_ITERS; ++it) {
        if (t < HW_) {                                 // row update: f
            const float* srow = &sim[t * HW_];
            float m = -1e30f;
            for (int j = 0; j < HW_; ++j) {
                float v = (gv[j] - 1.f + srow[j]) * INV_EPS;
                m = fmaxf(m, v);
            }
            float s = 0.f;
            for (int j = 0; j < HW_; ++j)
                s += __expf((gv[j] - 1.f + srow[j]) * INV_EPS - m);
            fv[t] = SINK_EPS * (la[t] - (m + __logf(s)));
        }
        __syncthreads();
        if (t < HW_) {                                 // col update: g
            float m = -1e30f;
            for (int i = 0; i < HW_; ++i) {
                float v = (fv[i] - 1.f + sim[i * HW_ + t]) * INV_EPS;
                m = fmaxf(m, v);
            }
            float s = 0.f;
            for (int i = 0; i < HW_; ++i)
                s += __expf((fv[i] - 1.f + sim[i * HW_ + t]) * INV_EPS - m);
            gv[t] = SINK_EPS * (lb[t] - (m + __logf(s)));
        }
        __syncthreads();
    }

    // --- Phase C: logits = sum(sim * plan) * T/n,  plan = exp((f+g-cost)/eps)*n ---
    float part = 0.f;
    for (int idx = t; idx < HW_ * HW_; idx += 128) {
        int i = idx >> 6, j = idx & 63;
        float s = sim[idx];
        float p = __expf((fv[i] + gv[j] - 1.f + s) * INV_EPS) * 64.f;
        part += s * p;
    }
    red[t] = part; __syncthreads();
    for (int st = 64; st > 0; st >>= 1) { if (t < st) red[t] += red[t + st]; __syncthreads(); }
    if (t == 0) out[pid] = red[0] * (TEMP_ / 64.0f);
}

// ---------------------------------------------------------------------------
extern "C" void kernel_launch(void* const* d_in, const int* in_sizes, int n_in,
                              void* d_out, int out_size, void* d_ws, size_t ws_size,
                              hipStream_t stream) {
    const float* query = (const float*)d_in[0];   // [8,75,640,64]
    const float* proto = (const float*)d_in[1];   // [8, 5,640,64]
    float* out = (float*)d_out;                   // [3000]

    char* ws = (char*)d_ws;
    size_t off = 0;
    auto carve = [&](size_t bytes) -> char* {
        char* p = ws + off;
        off = (off + bytes + 255) & ~(size_t)255;
        return p;
    };
    unsigned short* qn    = (unsigned short*)carve((size_t)B_ * Q_ * HW_ * C_ * 2); // 49.2 MB
    unsigned short* pnb   = (unsigned short*)carve((size_t)B_ * W_ * HW_ * C_ * 2); //  3.3 MB
    float*          qpool = (float*)carve((size_t)B_ * Q_ * C_ * 4);
    float*          ppool = (float*)carve((size_t)B_ * W_ * C_ * 4);
    float*          loga  = (float*)carve((size_t)NP_ * HW_ * 4);
    float*          logb  = (float*)carve((size_t)NP_ * HW_ * 4);

    norm_pool_kernel<<<B_ * Q_, 256, 0, stream>>>(query, qn, qpool);
    norm_pool_kernel<<<B_ * W_, 256, 0, stream>>>(proto, pnb, ppool);
    weights_kernel<<<NP_, 64, 0, stream>>>(qn, pnb, qpool, ppool, loga, logb);
    emd_kernel<<<NP_, 128, 0, stream>>>(qn, pnb, loga, logb, out);
}
